// MultiHeadLocalAttention_25168508354859
// MI455X (gfx1250) — compile-verified, matrix operand order checked by reading, not run on hardware
//
#include <hip/hip_runtime.h>
#include <stdint.h>

// ---------------------------------------------------------------------------
// MultiHeadLocalAttention for MI455X (gfx1250): wave32, WMMA f16->f32,
// TDM (tensor_load_to_lds) GEMM staging + async global->LDS in attention.
// ---------------------------------------------------------------------------

typedef _Float16 half_t;
typedef __attribute__((ext_vector_type(16))) _Float16 v16h;
typedef __attribute__((ext_vector_type(8)))  _Float16 v8h;
typedef __attribute__((ext_vector_type(8)))  float    v8f;
typedef __attribute__((ext_vector_type(4)))  uint32_t u32x4;
typedef __attribute__((ext_vector_type(8)))  uint32_t u32x8;

#define BATCH 4
#define SEQ   4096
#define DM    1024
#define HEADS 16
#define DKH   64
#define WINSZ 256
#define NWIN  (SEQ / WINSZ)
#define CTX   (2 * WINSZ)

__device__ __forceinline__ v8f wmma16(v16h a, v16h b, v8f c) {
  // D = A(16x32 f16) * B(32x16 f16) + C(16x16 f32)
  return __builtin_amdgcn_wmma_f32_16x16x32_f16(false, a, false, b, (short)0, c,
                                                false, false);
}

// Async copy 16 bytes global -> LDS (per lane). Tracked by ASYNCcnt.
// Generic LDS pointer low 32 bits == LDS byte offset (ISA 10.2 aperture map).
__device__ __forceinline__ void async_ld_b128(void* lds, const void* gaddr) {
  uint32_t l = (uint32_t)(uintptr_t)lds;
  asm volatile("global_load_async_to_lds_b128 %0, %1, off"
               :: "v"(l), "v"(gaddr) : "memory");
}
__device__ __forceinline__ void wait_async0() {
  asm volatile("s_wait_asynccnt 0x0" ::: "memory");
}

// Tensor Data Mover: DMA a 2D tile (tile_x f16 elements per row, tile_y rows,
// row stride stride_elems) from global memory into LDS, row-major packed.
// D# per ISA 8.3/8.4, 2-group form (2D tensor). Wave-level op, EXEC ignored,
// tracked by TENSORcnt. global_addr is the tile start (OOB origin = tile).
__device__ __forceinline__ void tdm_load_2d(void* lds, const void* g,
                                            uint32_t tile_x, uint32_t tile_y,
                                            uint32_t stride_elems) {
  const uint64_t ga = (uint64_t)(uintptr_t)g;
  u32x4 g0;
  g0[0] = 1u;                                              // count=1, load
  g0[1] = (uint32_t)(uintptr_t)lds;                        // lds_addr (bytes)
  g0[2] = (uint32_t)ga;                                    // global_addr[31:0]
  g0[3] = ((uint32_t)(ga >> 32) & 0x01FFFFFFu) | (2u << 30); // addr[56:32]|type=2
  u32x8 g1;
  g1[0] = 1u << 16;                                        // data_size=1 (2B)
  g1[1] = (tile_x & 0xFFFFu) << 16;                        // tensor_dim0[15:0]
  g1[2] = (tile_y & 0xFFFFu) << 16;                        // dim0[31:16]=0, dim1[15:0]
  g1[3] = (tile_x & 0xFFFFu) << 16;                        // dim1[31:16]=0, tile_dim0
  g1[4] = tile_y & 0xFFFFu;                                // tile_dim1, tile_dim2=0
  g1[5] = stride_elems;                                    // tensor_dim0_stride[31:0]
  g1[6] = 0u;                                              // stride0[47:32], stride1 lo
  g1[7] = 0u;                                              // stride1 hi
  asm volatile("tensor_load_to_lds %0, %1" :: "s"(g0), "s"(g1) : "memory");
}
__device__ __forceinline__ void wait_tensor0() {
  __builtin_amdgcn_s_wait_tensorcnt(0);
}

// Load a 16x32 f16 A/B fragment from LDS. Per ISA 7.12.2: lane half h gets
// K = {8h..8h+7} in v0..v3 and {16+8h..23+8h} in v4..v7 -> two contiguous
// 16-byte chunks at +8h and +16+8h halves within the row.
__device__ __forceinline__ v16h load_frag(const half_t* base, int row,
                                          int strideH, int kstep) {
  const int lane = threadIdx.x & 31;
  const int hf = lane >> 4;
  const half_t* p = base + row * strideH + kstep * 32 + hf * 8;
  v8h lo = *(const v8h*)(p);
  v8h hi = *(const v8h*)(p + 16);
  v16h r;
#pragma unroll
  for (int i = 0; i < 8; ++i) { r[i] = lo[i]; r[i + 8] = hi[i]; }
  return r;
}

// ---------------------------------------------------------------------------
// Elementwise f32 -> f16 conversion (memory-bound prologue pass).
// ---------------------------------------------------------------------------
__global__ __launch_bounds__(256) void cvt_f32_f16(const float* __restrict__ in,
                                                   half_t* __restrict__ out,
                                                   int n) {
  const int i = (blockIdx.x * 256 + threadIdx.x) * 8;
  if (i < n) {
#pragma unroll
    for (int j = 0; j < 8; ++j) out[i + j] = (half_t)in[i + j];
  }
}

// ---------------------------------------------------------------------------
// f16 GEMM: Y[m,n] = sum_k A[m,k]*B[n,k] + bias[n].
// Block tile 128x128, 8 waves of 32x64, K staged 32-wide. LDS double-buffered;
// each 128x32 tile staged by ONE TDM DMA issued by wave 0 before the current
// tile's WMMAs (latency hidden under compute), one barrier per K-step.
// MODE 0: (acc+bias)*scale -> f16 into (B,H,N,DK) layout (Q/K/V projections)
// MODE 1: acc+bias -> f32 row-major (final output projection)
// ---------------------------------------------------------------------------
template <int MODE>
__global__ __launch_bounds__(256) void gemm16_kernel(
    const half_t* __restrict__ Ah, const half_t* __restrict__ Bw,
    const float* __restrict__ bias, half_t* __restrict__ outH,
    float* __restrict__ outF, float scale) {
  __shared__ half_t lA[2][128 * 32];
  __shared__ half_t lB[2][128 * 32];

  const int m0 = blockIdx.y * 128;
  const int n0 = blockIdx.x * 128;
  const int t = threadIdx.x;
  const int wid = t >> 5, lane = t & 31, ln = lane & 15, hf = lane >> 4;
  const int wm = (wid >> 1) * 32;  // wave row offset (4 waves down)
  const int wn = (wid & 1) * 64;   // wave col offset (2 waves across)

  v8f acc[2][4];
#pragma unroll
  for (int i = 0; i < 2; ++i)
#pragma unroll
    for (int j = 0; j < 4; ++j)
#pragma unroll
      for (int e = 0; e < 8; ++e) acc[i][j][e] = 0.0f;

  const half_t* gA0 = Ah + (size_t)m0 * DM;   // tile origin, row stride DM
  const half_t* gB0 = Bw + (size_t)n0 * DM;
  const half_t* gAp = Ah + (size_t)(m0 + (t >> 1)) * DM + (t & 1) * 16;  // prefetch
  const half_t* gBp = Bw + (size_t)(n0 + (t >> 1)) * DM + (t & 1) * 16;

  // Prologue: wave 0 DMAs tile 0 into buffer 0 via TDM.
  if (wid == 0) {
    tdm_load_2d(&lA[0][0], gA0, 32, 128, DM);
    tdm_load_2d(&lB[0][0], gB0, 32, 128, DM);
    wait_tensor0();
  }
  __syncthreads();

  const int NK = DM / 32;
  for (int it = 0; it < NK; ++it) {
    const int cur = it & 1;
    // Wave 0 issues TDM staging for the next tile into the other buffer.
    if (wid == 0 && it + 1 < NK) {
      tdm_load_2d(&lA[cur ^ 1][0], gA0 + (it + 1) * 32, 32, 128, DM);
      tdm_load_2d(&lB[cur ^ 1][0], gB0 + (it + 1) * 32, 32, 128, DM);
    }
    // Per-lane L2 prefetch of the tile after next.
    if (it + 2 < NK) {
      __builtin_prefetch(gAp + (it + 2) * 32, 0, 0);
      __builtin_prefetch(gBp + (it + 2) * 32, 0, 0);
    }

    // Compute on the current buffer.
    v16h a[2], b[4];
#pragma unroll
    for (int mt = 0; mt < 2; ++mt) a[mt] = load_frag(lA[cur], wm + mt * 16 + ln, 32, 0);
#pragma unroll
    for (int nt = 0; nt < 4; ++nt) b[nt] = load_frag(lB[cur], wn + nt * 16 + ln, 32, 0);
#pragma unroll
    for (int mt = 0; mt < 2; ++mt)
#pragma unroll
      for (int nt = 0; nt < 4; ++nt) acc[mt][nt] = wmma16(a[mt], b[nt], acc[mt][nt]);

    if (wid == 0) wait_tensor0();  // next-tile DMA landed
    __syncthreads();               // publishes LDS; others done reading cur
  }

  float bv[4];
#pragma unroll
  for (int nt = 0; nt < 4; ++nt) bv[nt] = bias[n0 + wn + nt * 16 + ln];

#pragma unroll
  for (int mt = 0; mt < 2; ++mt) {
#pragma unroll
    for (int r = 0; r < 8; ++r) {
      const int m = m0 + wm + mt * 16 + r + 8 * hf;  // C layout: M = r + 8*half
#pragma unroll
      for (int nt = 0; nt < 4; ++nt) {
        const int nc = n0 + wn + nt * 16 + ln;       // C layout: N = lane%16
        if (MODE == 0) {
          const int bb = m >> 12, nn = m & (SEQ - 1);
          const int hh = nc >> 6, dd = nc & (DKH - 1);
          const float v = (acc[mt][nt][r] + bv[nt]) * scale;
          outH[(((size_t)(bb * HEADS + hh)) * SEQ + nn) * DKH + dd] = (half_t)v;
        } else {
          outF[(size_t)m * DM + nc] = acc[mt][nt][r] + bv[nt];
        }
      }
    }
  }
}

// ---------------------------------------------------------------------------
// Local attention with look-around: one workgroup per (batch*head, window).
// Flash-style online softmax over 16 key blocks of 32. Scale folded into Q.
// Q staged by one TDM DMA; K blocks via async global->LDS; V transposed.
// ---------------------------------------------------------------------------
__global__ __launch_bounds__(256) void local_attn_kernel(
    const half_t* __restrict__ Qh, const half_t* __restrict__ Kh,
    const half_t* __restrict__ Vh, const unsigned char* __restrict__ maskp,
    half_t* __restrict__ AO) {
  __shared__ half_t lQ[WINSZ * DKH];   // 32 KB: whole query window
  __shared__ half_t lK[32 * DKH];      // 4 KB: key block, key-major
  __shared__ half_t lVt[DKH * 32];     // 4 KB: value block, transposed
  __shared__ half_t lP[8][32 * 32];    // 16 KB: per-wave P tile

  const int blk = blockIdx.x;
  const int win = blk & (NWIN - 1);
  const int bh = blk >> 4;
  const int bb = bh >> 4;              // batch
  const int hh = bh & (HEADS - 1);     // head
  const int t = threadIdx.x, wid = t >> 5, lane = t & 31, ln = lane & 15, hf = lane >> 4;

  const half_t* Qb = Qh + ((size_t)bh * SEQ + (size_t)win * WINSZ) * DKH;
  const half_t* Kb = Kh + (size_t)bh * SEQ * DKH;
  const half_t* Vb = Vh + (size_t)bh * SEQ * DKH;

  // Wave 0 DMAs the whole contiguous 32KB query window (256*64 halves).
  if (wid == 0) tdm_load_2d(lQ, Qb, WINSZ * DKH, 1, WINSZ * DKH);

  const int qrow0 = wid * 32;          // this wave's query rows within window
  v8f Ox[2][4];
#pragma unroll
  for (int i = 0; i < 2; ++i)
#pragma unroll
    for (int j = 0; j < 4; ++j)
#pragma unroll
      for (int e = 0; e < 8; ++e) Ox[i][j][e] = 0.0f;

  float mrun[2][8], lrun[2][8];
#pragma unroll
  for (int i = 0; i < 2; ++i)
#pragma unroll
    for (int r = 0; r < 8; ++r) { mrun[i][r] = -3.0e38f; lrun[i][r] = 0.0f; }

  const int kstart = win * WINSZ - WINSZ;  // global key pos of context index 0
  const int krow = t >> 3;                 // 0..31 (key within block)
  const int kc = (t & 7) * 8;              // dk chunk base (16B)

  for (int kb = 0; kb < CTX / 32; ++kb) {
    __syncthreads();  // previous iteration's reads of lK/lVt complete
    const int kbase = kstart + kb * 32;
    // Validity is uniform per 32-key block: window 0, kb<8 is entirely padded.
    if (kbase >= 0) {
      const size_t krowg = (size_t)(kbase + krow) * DKH + kc;
      async_ld_b128(lK + krow * DKH + kc, Kb + krowg);
      v8h vv = *(const v8h*)(Vb + krowg);
#pragma unroll
      for (int i = 0; i < 8; ++i) lVt[(kc + i) * 32 + krow] = vv[i];
    } else {
      v8h z;
#pragma unroll
      for (int i = 0; i < 8; ++i) z[i] = (half_t)0.0f;
      *(v8h*)(lK + krow * DKH + kc) = z;
#pragma unroll
      for (int i = 0; i < 8; ++i) lVt[(kc + i) * 32 + krow] = (half_t)0.0f;
    }
    wait_async0();                  // this K block staged
    if (wid == 0) wait_tensor0();   // Q window staged (no-op after kb==0)
    __syncthreads();

    // S = Q(32x64) * Kblk^T(64x32): 2 m-tiles x 2 n-tiles x 2 k-steps.
    v8f s[2][2];
#pragma unroll
    for (int i = 0; i < 2; ++i)
#pragma unroll
      for (int j = 0; j < 2; ++j)
#pragma unroll
        for (int e = 0; e < 8; ++e) s[i][j][e] = 0.0f;

#pragma unroll
    for (int ks = 0; ks < 2; ++ks) {
      v16h a0 = load_frag(lQ, qrow0 + ln, DKH, ks);
      v16h a1 = load_frag(lQ, qrow0 + 16 + ln, DKH, ks);
      v16h b0 = load_frag(lK, ln, DKH, ks);
      v16h b1 = load_frag(lK, 16 + ln, DKH, ks);
      s[0][0] = wmma16(a0, b0, s[0][0]);
      s[0][1] = wmma16(a0, b1, s[0][1]);
      s[1][0] = wmma16(a1, b0, s[1][0]);
      s[1][1] = wmma16(a1, b1, s[1][1]);
    }

    // Key validity for this lane's two columns.
    int kp[2];
    bool km[2];
#pragma unroll
    for (int nt = 0; nt < 2; ++nt) {
      kp[nt] = kbase + nt * 16 + ln;
      const int safe = kp[nt] >= 0 ? kp[nt] : 0;
      km[nt] = (kp[nt] >= 0) && (maskp[(size_t)bb * SEQ + safe] != 0);
    }

    // Online softmax per row; xor 1/2/4/8 reductions stay within a 16-lane half.
#pragma unroll
    for (int mt = 0; mt < 2; ++mt) {
#pragma unroll
      for (int r = 0; r < 8; ++r) {
        const int qpos = win * WINSZ + qrow0 + mt * 16 + r + 8 * hf;
        const bool ok0 = km[0] && (kp[0] <= qpos);
        const bool ok1 = km[1] && (kp[1] <= qpos);
        const float s0 = ok0 ? s[mt][0][r] : -1.0e30f;
        const float s1 = ok1 ? s[mt][1][r] : -1.0e30f;
        float mx = fmaxf(s0, s1);
#pragma unroll
        for (int d = 8; d >= 1; d >>= 1) mx = fmaxf(mx, __shfl_xor(mx, d, 32));
        const float mnew = fmaxf(mrun[mt][r], mx);
        const float alpha = __expf(mrun[mt][r] - mnew);
        const float p0 = ok0 ? __expf(s0 - mnew) : 0.0f;  // masked -> exactly 0
        const float p1 = ok1 ? __expf(s1 - mnew) : 0.0f;
        float ps = p0 + p1;
#pragma unroll
        for (int d = 8; d >= 1; d >>= 1) ps += __shfl_xor(ps, d, 32);
        lrun[mt][r] = lrun[mt][r] * alpha + ps;
        mrun[mt][r] = mnew;
#pragma unroll
        for (int dt = 0; dt < 4; ++dt) Ox[mt][dt][r] *= alpha;
        // Scatter P into wave-private LDS tile for A-fragment reload.
        const int prow = mt * 16 + r + 8 * hf;
        lP[wid][prow * 32 + ln] = (half_t)p0;
        lP[wid][prow * 32 + 16 + ln] = (half_t)p1;
      }
    }

    // O += P(32x32) * V(32x64): 2 m-tiles x 4 dk-tiles.
    v16h pa0 = load_frag(lP[wid], ln, 32, 0);
    v16h pa1 = load_frag(lP[wid], 16 + ln, 32, 0);
#pragma unroll
    for (int dt = 0; dt < 4; ++dt) {
      v16h bv = load_frag(lVt, dt * 16 + ln, 32, 0);
      Ox[0][dt] = wmma16(pa0, bv, Ox[0][dt]);
      Ox[1][dt] = wmma16(pa1, bv, Ox[1][dt]);
    }
  }

  // Normalize and store to (B, N, H*DK) f16 for the output projection.
#pragma unroll
  for (int mt = 0; mt < 2; ++mt) {
#pragma unroll
    for (int r = 0; r < 8; ++r) {
      const float inv = 1.0f / lrun[mt][r];
      const int q = win * WINSZ + qrow0 + mt * 16 + r + 8 * hf;
      const size_t rb = ((size_t)bb * SEQ + q) * DM + (size_t)hh * DKH;
#pragma unroll
      for (int dt = 0; dt < 4; ++dt)
        AO[rb + dt * 16 + ln] = (half_t)(Ox[mt][dt][r] * inv);
    }
  }
}

// ---------------------------------------------------------------------------
extern "C" void kernel_launch(void* const* d_in, const int* in_sizes, int n_in,
                              void* d_out, int out_size, void* d_ws, size_t ws_size,
                              hipStream_t stream) {
  const float* query = (const float*)d_in[0];
  const float* key   = (const float*)d_in[1];
  const float* value = (const float*)d_in[2];
  const unsigned char* mask = (const unsigned char*)d_in[3];
  const float* W_q = (const float*)d_in[4];
  const float* b_q = (const float*)d_in[5];
  const float* W_k = (const float*)d_in[6];
  const float* b_k = (const float*)d_in[7];
  const float* W_v = (const float*)d_in[8];
  const float* b_v = (const float*)d_in[9];
  const float* W_o = (const float*)d_in[10];
  const float* b_o = (const float*)d_in[11];

  const size_t ACT = (size_t)BATCH * SEQ * DM;  // 16,777,216 elements
  const size_t WEL = (size_t)DM * DM;           //  1,048,576 elements

  half_t* p = (half_t*)d_ws;
  half_t* Xq = p; p += ACT;   // f16 activations
  half_t* Xk = p; p += ACT;
  half_t* Xv = p; p += ACT;
  half_t* Wqh = p; p += WEL;  // f16 weights
  half_t* Wkh = p; p += WEL;
  half_t* Wvh = p; p += WEL;
  half_t* Woh = p; p += WEL;
  half_t* Qw = p; p += ACT;   // projected Q/K/V, (B,H,N,DK)
  half_t* Kw = p; p += ACT;
  half_t* Vw = p; p += ACT;
  half_t* AOw = Xq;           // reuse: Xq dead after Q projection

  // f32 -> f16 conversion passes (memory-bound, ~13us at 23.3 TB/s).
  cvt_f32_f16<<<ACT / (8 * 256), 256, 0, stream>>>(query, Xq, (int)ACT);
  cvt_f32_f16<<<ACT / (8 * 256), 256, 0, stream>>>(key,   Xk, (int)ACT);
  cvt_f32_f16<<<ACT / (8 * 256), 256, 0, stream>>>(value, Xv, (int)ACT);
  cvt_f32_f16<<<WEL / (8 * 256), 256, 0, stream>>>(W_q, Wqh, (int)WEL);
  cvt_f32_f16<<<WEL / (8 * 256), 256, 0, stream>>>(W_k, Wkh, (int)WEL);
  cvt_f32_f16<<<WEL / (8 * 256), 256, 0, stream>>>(W_v, Wvh, (int)WEL);
  cvt_f32_f16<<<WEL / (8 * 256), 256, 0, stream>>>(W_o, Woh, (int)WEL);

  const dim3 gg(DM / 128, (BATCH * SEQ) / 128);  // (8, 128)
  const float qscale = 0.125f;                   // DK^-0.5 = 64^-0.5

  gemm16_kernel<0><<<gg, 256, 0, stream>>>(Xq, Wqh, b_q, Qw, nullptr, qscale);
  gemm16_kernel<0><<<gg, 256, 0, stream>>>(Xk, Wkh, b_k, Kw, nullptr, 1.0f);
  gemm16_kernel<0><<<gg, 256, 0, stream>>>(Xv, Wvh, b_v, Vw, nullptr, 1.0f);

  local_attn_kernel<<<BATCH * HEADS * NWIN, 256, 0, stream>>>(Qw, Kw, Vw, mask, AOw);

  gemm16_kernel<1><<<gg, 256, 0, stream>>>(AOw, Woh, b_o, nullptr, (float*)d_out, 1.0f);
}
